// GAT_11106785427688
// MI455X (gfx1250) — matrix operand order repaired and checked by reading, compiled, not run on hardware
//
#include <hip/hip_runtime.h>

// GAT 2-layer forward for MI455X (gfx1250, wave32).
//   1) gat_gemm64_wmma : h = x@W via v_wmma_f32_16x16x32_f16; W staged to LDS by the
//                        Tensor Data Mover (tensor_load_to_lds + s_wait_tensorcnt),
//                        then converted f32 -> f16-transposed in LDS.
//   2) gat_alpha       : per-node attention logits
//   3) gat_zero        : clear [m | denom | acc]
//   4) gat_edge_max    : ordered-u32 atomicMax segment-max of leakyrelu(e)
//   5) gat_edge_acc    : fused: w = exp(e - m[dst]); denom[dst]+=w; acc[dst]+=w*h[src]
//   6) finish          : out = acc/denom (+bias, ELU) ; layer2: + wave-level L2 normalize

#define NEG_SLOPE 0.2f

typedef __attribute__((ext_vector_type(16))) _Float16     v16h;
typedef __attribute__((ext_vector_type(8)))  float        v8f;
typedef __attribute__((ext_vector_type(4)))  unsigned int v4u;
typedef __attribute__((ext_vector_type(8)))  int          v8i;
typedef __attribute__((ext_vector_type(4)))  int          v4i;

#if __has_builtin(__builtin_amdgcn_tensor_load_to_lds)
#define GAT_USE_TDM 1
#else
#define GAT_USE_TDM 0
#endif

__device__ __forceinline__ unsigned f2ord(float f) {
    unsigned u = __float_as_uint(f);
    return (u & 0x80000000u) ? ~u : (u | 0x80000000u);
}
__device__ __forceinline__ float ord2f(unsigned o) {
    return (o & 0x80000000u) ? __uint_as_float(o & 0x7FFFFFFFu) : __uint_as_float(~o);
}

// ---------------- Dense projection: H[N,64] = X[N,64] @ W[64,64] ----------------
// One wave -> 16 rows x 64 cols. 4 N-tiles x 2 K-tiles = 8 WMMAs per wave.
struct GemmSmem {
    float Wstage[64 * 64];                  // 16 KB; TDM target (first LDS object -> offset 0)
    __align__(32) _Float16 WT[64 * 64];     // 8 KB; transposed f16: WT[col*64 + k]
};

__global__ __launch_bounds__(256) void gat_gemm64_wmma(const float* __restrict__ X,
                                                       const float* __restrict__ W,
                                                       float* __restrict__ H, int N)
{
    __shared__ GemmSmem sm;
    int t = threadIdx.x;

#if GAT_USE_TDM
    // ---- Tensor Data Mover: DMA W (64x64 f32) global -> LDS staging ----
    if (t < 32) {   // single wave issues the descriptor (TDM ignores EXEC)
        unsigned long long ga = (unsigned long long)W;
        // D# group0: count=1 | lds_addr | global_addr[56:0] | type=2
        v4u g0 = { 1u,
                   0u /* LDS offset of sm.Wstage (first LDS object) */,
                   (unsigned)ga,
                   (unsigned)((ga >> 32) & 0x1FFFFFFu) | (2u << 30) };
        // D# group1: data_size=2 (4B); tensor_dim0=64 @bits[79:48]; tensor_dim1=64 @[111:80];
        //            tile_dim0=64 @[127:112]; tile_dim1=64 @[143:128]; stride0=64 @[207:160]
        v8i g1 = { (int)(2u << 16),
                   (int)(64u << 16),
                   (int)(64u << 16),
                   (int)(64u << 16),
                   64,
                   64,
                   0,
                   0 };
        v4i g2 = { 0, 0, 0, 0 };            // 2-D tile: groups 2/3 unused
        v4i g3 = { 0, 0, 0, 0 };
        v8i g4 = { 0, 0, 0, 0, 0, 0, 0, 0 };// trailing group (clang-23 6-arg form)
        __builtin_amdgcn_tensor_load_to_lds(g0, g1, g2, g3, g4, 0);
        __builtin_amdgcn_s_wait_tensorcnt(0);
    }
    __syncthreads();
    const float* Wsrc = sm.Wstage;
#else
    const float* Wsrc = W;
#endif

    // Convert to f16, transposed, cooperatively (4096 elems / 256 threads)
    #pragma unroll
    for (int i = 0; i < 16; ++i) {
        int g = t * 16 + i;
        int k = g >> 6, c = g & 63;
        sm.WT[c * 64 + k] = (_Float16)Wsrc[g];
    }
    __syncthreads();

    int wave   = t >> 5;
    int lane   = t & 31;
    int halfid = lane >> 4;
    int mrow   = lane & 15;
    int r0     = blockIdx.x * 128 + wave * 16;
    int row    = r0 + mrow; if (row >= N) row = N - 1;   // clamp loads; stores guarded
    const float* xr = X + (size_t)row * 64;

    // A fragments (contiguous k order, B uses the same): lanes<16 hold K{b..b+15}, lanes>=16 hold K{b+16..b+31}.
    v16h a[2];
    #pragma unroll
    for (int kt = 0; kt < 2; ++kt) {
        int kb = kt * 32 + halfid * 16;
        float4 f0 = *(const float4*)(xr + kb);
        float4 f1 = *(const float4*)(xr + kb + 4);
        float4 f2 = *(const float4*)(xr + kb + 8);
        float4 f3 = *(const float4*)(xr + kb + 12);
        v16h av;
        av[0]=(_Float16)f0.x; av[1]=(_Float16)f0.y; av[2]=(_Float16)f0.z; av[3]=(_Float16)f0.w;
        av[4]=(_Float16)f1.x; av[5]=(_Float16)f1.y; av[6]=(_Float16)f1.z; av[7]=(_Float16)f1.w;
        av[8]=(_Float16)f2.x; av[9]=(_Float16)f2.y; av[10]=(_Float16)f2.z; av[11]=(_Float16)f2.w;
        av[12]=(_Float16)f3.x; av[13]=(_Float16)f3.y; av[14]=(_Float16)f3.z; av[15]=(_Float16)f3.w;
        a[kt] = av;
    }

    int  rbase    = r0 + halfid * 8;
    bool fullTile = (rbase + 8 <= N);
    #pragma unroll
    for (int nt = 0; nt < 4; ++nt) {
        int col = nt * 16 + mrow;
        const v16h* b0 = (const v16h*)(sm.WT + col * 64 + halfid * 16);
        const v16h* b1 = (const v16h*)(sm.WT + col * 64 + 32 + halfid * 16);
        v8f acc = {0.f, 0.f, 0.f, 0.f, 0.f, 0.f, 0.f, 0.f};
        acc = __builtin_amdgcn_wmma_f32_16x16x32_f16(false, a[0], false, *b0, (short)0, acc, false, false);
        acc = __builtin_amdgcn_wmma_f32_16x16x32_f16(false, a[1], false, *b1, (short)0, acc, false, false);
        float* hp = H + (size_t)rbase * 64 + col;      // single 64-bit base, imm offsets
        if (fullTile) {
            #pragma unroll
            for (int j = 0; j < 8; ++j) hp[j * 64] = acc[j];
        } else {
            #pragma unroll
            for (int j = 0; j < 8; ++j) if (rbase + j < N) hp[j * 64] = acc[j];
        }
    }
}

// ---------------- Per-node attention logits ----------------
template <int HN, int C>
__global__ void gat_alpha(const float* __restrict__ Hf, const float* __restrict__ a_s,
                          const float* __restrict__ a_d, float* __restrict__ as,
                          float* __restrict__ ad, int N)
{
    int idx = blockIdx.x * blockDim.x + threadIdx.x;
    if (idx >= N * HN) return;
    int n = idx / HN, hd = idx % HN;
    const float* hr = Hf + (size_t)n * 64 + hd * C;
    float ss = 0.f, sd = 0.f;
    #pragma unroll
    for (int c = 0; c < C; ++c) { float v = hr[c]; ss += v * a_s[hd * C + c]; sd += v * a_d[hd * C + c]; }
    as[idx] = ss; ad[idx] = sd;
}

__global__ void gat_zero(float* __restrict__ p, long long n)
{
    long long i = (long long)blockIdx.x * blockDim.x + threadIdx.x;
    long long s = (long long)gridDim.x * blockDim.x;
    for (; i < n; i += s) p[i] = 0.f;
}

// ---------------- Edge pass 1: segment-max of leakyrelu(e) ----------------
template <int HN>
__global__ void gat_edge_max(const int* __restrict__ ei, const float* __restrict__ as,
                             const float* __restrict__ ad, unsigned* __restrict__ m,
                             int E, int N)
{
    long long idx = (long long)blockIdx.x * blockDim.x + threadIdx.x;
    if (idx >= (long long)(E + N) * HN) return;
    int e = (int)(idx / HN), hd = (int)(idx % HN);
    int src, dst;
    if (e < E) { src = ei[e]; dst = ei[E + e]; } else { src = e - E; dst = src; }   // self loop
    float el = as[src * HN + hd] + ad[dst * HN + hd];
    el = el > 0.f ? el : NEG_SLOPE * el;
    atomicMax(&m[dst * HN + hd], f2ord(el));
}

// ---------------- Edge pass 2 (fused): denom += w ; acc += w * h[src] ----------------
template <int HN>
__global__ void gat_edge_acc(const int* __restrict__ ei, const float* __restrict__ as,
                             const float* __restrict__ ad, const unsigned* __restrict__ m,
                             float* __restrict__ denom, const float* __restrict__ Hf,
                             float* __restrict__ acc, int E, int N)
{
    long long idx = (long long)blockIdx.x * blockDim.x + threadIdx.x;
    if (idx >= (long long)(E + N) * 64) return;
    int e = (int)(idx >> 6), c = (int)(idx & 63);
    const int C = 64 / HN;
    int hd = (HN == 1) ? 0 : (c >> 4);
    int src, dst;
    if (e < E) { src = ei[e]; dst = ei[E + e]; } else { src = e - E; dst = src; }
    float el = as[src * HN + hd] + ad[dst * HN + hd];
    el = el > 0.f ? el : NEG_SLOPE * el;
    float w = __expf(el - ord2f(m[dst * HN + hd]));
    if ((c & (C - 1)) == 0) atomicAdd(&denom[dst * HN + hd], w);
    atomicAdd(&acc[(size_t)dst * 64 + c], w * Hf[(size_t)src * 64 + c]);
}

// ---------------- Layer-1 epilogue: normalize, bias, ELU ----------------
__global__ void gat_finish1(const float* __restrict__ acc, const float* __restrict__ denom,
                            const float* __restrict__ b1, float* __restrict__ xo, int N)
{
    int idx = blockIdx.x * blockDim.x + threadIdx.x;
    if (idx >= N * 64) return;
    int n = idx >> 6, c = idx & 63;
    float v = acc[idx] / (denom[n * 4 + (c >> 4)] + 1e-16f) + b1[c];
    xo[idx] = v > 0.f ? v : (__expf(v) - 1.f);
}

// ---------------- Layer-2 epilogue: normalize, bias, L2 row-normalize (wave/node) ----------------
__global__ __launch_bounds__(256) void gat_finish2(const float* __restrict__ acc,
                                                   const float* __restrict__ denom,
                                                   const float* __restrict__ b2,
                                                   float* __restrict__ out, int N)
{
    int wave = threadIdx.x >> 5, lane = threadIdx.x & 31;
    int n = blockIdx.x * 8 + wave;
    if (n >= N) return;
    float d  = denom[n] + 1e-16f;
    float v0 = acc[(size_t)n * 64 + lane]      / d + b2[lane];
    float v1 = acc[(size_t)n * 64 + 32 + lane] / d + b2[32 + lane];
    float ss = v0 * v0 + v1 * v1;
    #pragma unroll
    for (int off = 16; off; off >>= 1) ss += __shfl_xor(ss, off, 32);   // wave32 butterfly
    float nrm = sqrtf(ss); nrm = nrm > 1e-12f ? nrm : 1e-12f;
    out[(size_t)n * 64 + lane]      = v0 / nrm;
    out[(size_t)n * 64 + 32 + lane] = v1 / nrm;
}

extern "C" void kernel_launch(void* const* d_in, const int* in_sizes, int n_in,
                              void* d_out, int out_size, void* d_ws, size_t ws_size,
                              hipStream_t stream)
{
    const int*   ei  = (const int*)d_in[0];     // edge_index [2,E]
    const float* emb = (const float*)d_in[1];   // [N,64]
    const float* W1  = (const float*)d_in[2];
    const float* as1 = (const float*)d_in[3];
    const float* ad1 = (const float*)d_in[4];
    const float* b1  = (const float*)d_in[5];
    const float* W2  = (const float*)d_in[6];
    const float* as2 = (const float*)d_in[7];
    const float* ad2 = (const float*)d_in[8];
    const float* b2  = (const float*)d_in[9];
    int E = in_sizes[0] / 2;
    int N = in_sizes[1] / 64;

    // Workspace layout (floats): h | xbuf | as | ad | m | denom | acc  = 208*N floats (~83 MB)
    float* h    = (float*)d_ws;
    float* xbuf = h    + (size_t)N * 64;
    float* asb  = xbuf + (size_t)N * 64;
    float* adb  = asb  + (size_t)N * 4;
    float* mbuf = adb  + (size_t)N * 4;
    float* den  = mbuf + (size_t)N * 4;
    float* acc  = den  + (size_t)N * 4;

    long long Etot = (long long)E + N;
    dim3 blk(256);
    int gemmBlocks = (N + 127) / 128;
    long long zn = (long long)N * 72;            // m + denom + acc are contiguous

    // ---- Layer 1: GATConv(64 -> 16, heads=4, concat) ----
    gat_gemm64_wmma<<<gemmBlocks, blk, 0, stream>>>(emb, W1, h, N);
    gat_alpha<4, 16><<<(N * 4 + 255) / 256, blk, 0, stream>>>(h, as1, ad1, asb, adb, N);
    gat_zero<<<2048, blk, 0, stream>>>(mbuf, zn);
    gat_edge_max<4><<<(unsigned)((Etot * 4 + 255) / 256), blk, 0, stream>>>(ei, asb, adb, (unsigned*)mbuf, E, N);
    gat_edge_acc<4><<<(unsigned)((Etot * 64 + 255) / 256), blk, 0, stream>>>(ei, asb, adb, (const unsigned*)mbuf, den, h, acc, E, N);
    gat_finish1<<<(N * 64 + 255) / 256, blk, 0, stream>>>(acc, den, b1, xbuf, N);

    // ---- Layer 2: GATConv(64 -> 64, heads=1) + L2 normalize ----
    gat_gemm64_wmma<<<gemmBlocks, blk, 0, stream>>>(xbuf, W2, h, N);
    gat_alpha<1, 64><<<(N + 255) / 256, blk, 0, stream>>>(h, as2, ad2, asb, adb, N);
    gat_zero<<<2048, blk, 0, stream>>>(mbuf, zn);
    gat_edge_max<1><<<(unsigned)((Etot + 255) / 256), blk, 0, stream>>>(ei, asb, adb, (unsigned*)mbuf, E, N);
    gat_edge_acc<1><<<(unsigned)((Etot * 64 + 255) / 256), blk, 0, stream>>>(ei, asb, adb, (const unsigned*)mbuf, den, h, acc, E, N);
    gat_finish2<<<(N + 7) / 8, blk, 0, stream>>>(acc, den, b2, (float*)d_out, N);
}
